// MH_ATT_90975997264665
// MI455X (gfx1250) — matrix operand order repaired and checked by reading, not compiled, not run on hardware
//
#include <hip/hip_runtime.h>
#include <hip/hip_bf16.h>
#include <stdint.h>

#define NQc 4096
#define NKVc 4096
#define NBc 16
#define Hc 4
#define DKc 32
#define DIMc 128

typedef __attribute__((ext_vector_type(16))) __bf16 bf16x16;
typedef __attribute__((ext_vector_type(8)))  float  f32x8;

struct BPair { uint4 lo, hi; };

__device__ __forceinline__ uint16_t f2bf_bits(float f) {
  uint32_t u = __builtin_bit_cast(uint32_t, f);
  u += 0x7FFFu + ((u >> 16) & 1u);             // round-to-nearest-even
  return (uint16_t)(u >> 16);
}

__device__ __forceinline__ __bf16 f2bf(float f) {
  uint16_t h = f2bf_bits(f);
  return __builtin_bit_cast(__bf16, h);
}

__device__ __forceinline__ f32x8 zero8() {
  f32x8 z;
#pragma unroll
  for (int i = 0; i < 8; ++i) z[i] = 0.0f;
  return z;
}

__device__ __forceinline__ f32x8 wmma_bf16(bf16x16 a, bf16x16 b, f32x8 c) {
  // D = A(16x32 bf16) * B(32x16 bf16) + C(16x16 f32)
  return __builtin_amdgcn_wmma_f32_16x16x32_bf16(
      /*neg_a=*/false, a, /*neg_b=*/false, b,
      /*c_mod=*/(short)0, c, /*reuse_a=*/false, /*reuse_b=*/false);
}

// 8 bf16 from p0 and 8 bf16 from p1 (each 16B-aligned) -> one bf16x16 operand
__device__ __forceinline__ bf16x16 ld16(const __bf16* p0, const __bf16* p1) {
  BPair t;
  t.lo = *(const uint4*)p0;
  t.hi = *(const uint4*)p1;
  return __builtin_bit_cast(bf16x16, t);
}

// -------- bulk fp32 -> bf16 (bandwidth-bound, L2 resident)
__global__ __launch_bounds__(256) void cvt_kernel(const float* __restrict__ in,
                                                  __bf16* __restrict__ out,
                                                  int n4) {
  int i = blockIdx.x * blockDim.x + threadIdx.x;
  if (i < n4) {
    float4 v = ((const float4*)in)[i];
    uint2 p;
    p.x = (uint32_t)f2bf_bits(v.x) | ((uint32_t)f2bf_bits(v.y) << 16);
    p.y = (uint32_t)f2bf_bits(v.z) | ((uint32_t)f2bf_bits(v.w) << 16);
    ((uint2*)out)[i] = p;
  }
}

// -------- segment boundaries from sorted batch ids: seg[b] = first idx with id>=b
__global__ void seg_kernel(const int* __restrict__ ids, int* __restrict__ seg,
                           int n, int nb) {
  int i = blockIdx.x * blockDim.x + threadIdx.x;
  if (i == 0) { seg[0] = 0; seg[nb] = n; }
  if (i > 0 && i < n) {
    int a = ids[i - 1], b = ids[i];
    for (int g = a + 1; g <= b; ++g) seg[g] = i;
  }
}

// -------- C[M,128] = X[M,128] @ W[128,128]^T + bias, pure-bf16 operands.
// Two M-tiles per wave (each B tile feeds 2 WMMAs) + software-pipelined B.
// OutMode 0: bf16 row-major [row][128]
// OutMode 1: bf16 transposed [col][ldt]   (for V projection)
// OutMode 2: f32  row-major [row][128]    (final output)
template <int OutMode>
__global__ __launch_bounds__(256) void gemm128_wmma(
    const __bf16* __restrict__ X, const __bf16* __restrict__ W,
    const float* __restrict__ bias, void* __restrict__ outp, int ldt) {
  const int t = threadIdx.x;
  const int w = t >> 5;
  const int L = t & 31;
  const int lane16 = L & 15;
  const int half = L >> 4;
  const int row0 = blockIdx.x * 256;           // 8 waves x 32 rows
  const int ka0 = half ? 8 : 0;                // A: K {0-7,16-23} / {8-15,24-31}
  const int kb0 = half ? 8 : 0;                // B: K {0-7,16-23} / {8-15,24-31}
  const int arow0 = row0 + w * 32 + lane16;    // M-tile 0
  const int arow1 = arow0 + 16;                // M-tile 1

  f32x8 acc0[8], acc1[8];
#pragma unroll
  for (int n = 0; n < 8; ++n) { acc0[n] = zero8(); acc1[n] = zero8(); }

#pragma unroll
  for (int kc = 0; kc < 4; ++kc) {
    const __bf16* a0p = X + arow0 * DIMc + kc * 32 + ka0;
    const __bf16* a1p = X + arow1 * DIMc + kc * 32 + ka0;
    const bf16x16 a0 = ld16(a0p, a0p + 16);
    const bf16x16 a1 = ld16(a1p, a1p + 16);
    const __bf16* wb = W + lane16 * DIMc + kc * 32 + kb0;
    bf16x16 bcur = ld16(wb, wb + 16);          // n = 0 tile
#pragma unroll
    for (int n = 0; n < 8; ++n) {
      bf16x16 bnext = bcur;
      if (n < 7) {                             // prefetch B tile n+1
        const __bf16* bp = wb + (n + 1) * 16 * DIMc;
        bnext = ld16(bp, bp + 16);
      }
      acc0[n] = wmma_bf16(a0, bcur, acc0[n]);
      acc1[n] = wmma_bf16(a1, bcur, acc1[n]);
      bcur = bnext;
    }
  }

  // C layout: VGPR j -> rows j / j+8 per half-wave; N = lane16
#pragma unroll
  for (int m = 0; m < 2; ++m) {
    const int rbase = row0 + w * 32 + m * 16 + half * 8;
#pragma unroll
    for (int n = 0; n < 8; ++n) {
      const int col = n * 16 + lane16;
      const float bc = bias[col];
#pragma unroll
      for (int j = 0; j < 8; ++j) {
        const int row = rbase + j;
        const float v = (m == 0 ? acc0[n][j] : acc1[n][j]) + bc;
        if (OutMode == 0)      ((__bf16*)outp)[row * DIMc + col] = f2bf(v);
        else if (OutMode == 1) ((__bf16*)outp)[col * ldt + row]  = f2bf(v);
        else                   ((float*)outp)[row * DIMc + col]  = v;
      }
    }
  }
}

// -------- segment-masked flash attention: 16 queries/block, wave = head
__global__ __launch_bounds__(128) void attn_wmma(
    const __bf16* __restrict__ qp,   // [NQ][128] bf16
    const __bf16* __restrict__ kp,   // [NKV][128] bf16
    const __bf16* __restrict__ vpT,  // [128][NKV] bf16 (d-major)
    const int* __restrict__ qb, const int* __restrict__ kvseg,
    const int* __restrict__ kvb,
    __bf16* __restrict__ att)        // [NQ][128] bf16
{
  __shared__ float S[Hc][16][33];
  __shared__ float alpha_s[Hc][16];
  __shared__ float linv_s[Hc][16];
  __shared__ int kvid_s[32];
  __shared__ int qb_s[16];

  const int t = threadIdx.x;
  const int w = t >> 5;        // head index
  const int L = t & 31;
  const int lane16 = L & 15;
  const int half = L >> 4;
  const int q0 = blockIdx.x * 16;

  if (t < 16) qb_s[t] = qb[q0 + t];
  __syncthreads();
  const int kv_lo = kvseg[qb_s[0]];
  const int kv_hi = kvseg[qb_s[15] + 1];

  // A operand: Q tile (16 queries x DK=32) for this head, loaded once
  const int qrow = q0 + lane16;
  const __bf16* aptr = qp + qrow * DIMc + w * DKc + (half ? 8 : 0);
  const bf16x16 aq = ld16(aptr, aptr + 16);

  f32x8 o0 = zero8(), o1 = zero8();
  float m_i = -1e30f, l_i = 0.0f;
  const float scale = 0.17677669529663687f;  // 1/sqrt(32)
  const int rb = half * 8;

  for (int kb = kv_lo & ~31; kb < kv_hi; kb += 32) {
    if (t < 32) kvid_s[t] = kvb[kb + t];

    // S = Q * K^T  (two 16-kv tiles; K-dim = DK = 32 exactly)
    const int dK = w * DKc + (half ? 8 : 0);
    const __bf16* k0p = kp + (kb + lane16) * DIMc + dK;
    const __bf16* k1p = kp + (kb + 16 + lane16) * DIMc + dK;
    const bf16x16 b0 = ld16(k0p, k0p + 16);
    const bf16x16 b1 = ld16(k1p, k1p + 16);
    f32x8 s0 = wmma_bf16(aq, b0, zero8());
    f32x8 s1 = wmma_bf16(aq, b1, zero8());
#pragma unroll
    for (int j = 0; j < 8; ++j) {
      S[w][rb + j][lane16]      = s0[j];
      S[w][rb + j][16 + lane16] = s1[j];
    }
    __syncthreads();

    // online softmax, one row per lane 0..15 (exact batch-id masking)
    if (L < 16) {
      const int row = L;
      const int g = qb_s[row];
      float mnew = m_i;
#pragma unroll
      for (int c = 0; c < 32; ++c)
        if (kvid_s[c] == g) mnew = fmaxf(mnew, S[w][row][c] * scale);
      const float al = __expf(m_i - mnew);
      float ls = 0.0f;
#pragma unroll
      for (int c = 0; c < 32; ++c) {
        float p = 0.0f;
        if (kvid_s[c] == g) p = __expf(S[w][row][c] * scale - mnew);
        ls += p;
        S[w][row][c] = p;
      }
      l_i = l_i * al + ls;
      m_i = mnew;
      alpha_s[w][row] = al;
    }
    __syncthreads();

    // rescale running O by alpha
#pragma unroll
    for (int j = 0; j < 8; ++j) {
      const float al = alpha_s[w][rb + j];
      o0[j] *= al;
      o1[j] *= al;
    }

    // P in A-layout (row = lane16; K split by half-wave), from LDS
    const int k0 = half ? 8 : 0;
    bf16x16 pm;
#pragma unroll
    for (int i = 0; i < 8; ++i) {
      pm[i]     = f2bf(S[w][lane16][k0 + i]);
      pm[8 + i] = f2bf(S[w][lane16][k0 + 16 + i]);
    }

    // O += P * V  (V transposed layout -> contiguous per-lane kv runs)
    const int kvoff = kb + (half ? 8 : 0);
    const __bf16* v0p = vpT + (w * DKc + lane16) * NKVc + kvoff;
    const __bf16* v1p = vpT + (w * DKc + 16 + lane16) * NKVc + kvoff;
    const bf16x16 bv0 = ld16(v0p, v0p + 16);
    const bf16x16 bv1 = ld16(v1p, v1p + 16);
    o0 = wmma_bf16(pm, bv0, o0);
    o1 = wmma_bf16(pm, bv1, o1);
    __syncthreads();
  }

  if (L < 16) linv_s[w][L] = (l_i > 0.0f) ? (1.0f / l_i) : 0.0f;
  __syncthreads();
#pragma unroll
  for (int j = 0; j < 8; ++j) {
    const float li = linv_s[w][rb + j];
    const int row = q0 + rb + j;
    att[row * DIMc + w * DKc + lane16]      = f2bf(o0[j] * li);
    att[row * DIMc + w * DKc + 16 + lane16] = f2bf(o1[j] * li);
  }
}

extern "C" void kernel_launch(void* const* d_in, const int* in_sizes, int n_in,
                              void* d_out, int out_size, void* d_ws, size_t ws_size,
                              hipStream_t stream) {
  (void)in_sizes; (void)n_in; (void)out_size; (void)ws_size;
  const float* query = (const float*)d_in[0];
  const float* key   = (const float*)d_in[1];
  const float* value = (const float*)d_in[2];
  const float* Wq = (const float*)d_in[3];
  const float* bq = (const float*)d_in[4];
  const float* Wk = (const float*)d_in[5];
  const float* bk = (const float*)d_in[6];
  const float* Wv = (const float*)d_in[7];
  const float* bv = (const float*)d_in[8];
  const float* Wo = (const float*)d_in[9];
  const float* bo = (const float*)d_in[10];
  const int* qb  = (const int*)d_in[11];
  const int* kvb = (const int*)d_in[12];

  const size_t MB = 1u << 20;
  char* ws = (char*)d_ws;
  __bf16* Xq  = (__bf16*)(ws + 0 * MB);        // 1 MiB each
  __bf16* Xk  = (__bf16*)(ws + 1 * MB);
  __bf16* Xv  = (__bf16*)(ws + 2 * MB);
  __bf16* qp  = (__bf16*)(ws + 3 * MB);
  __bf16* kp  = (__bf16*)(ws + 4 * MB);
  __bf16* vpT = (__bf16*)(ws + 5 * MB);        // transposed [128][NKV]
  __bf16* atb = (__bf16*)(ws + 6 * MB);
  __bf16* Wqb = (__bf16*)(ws + 7 * MB);        // 32 KiB each
  __bf16* Wkb = (__bf16*)(ws + 7 * MB + 64 * 1024);
  __bf16* Wvb = (__bf16*)(ws + 7 * MB + 128 * 1024);
  __bf16* Wob = (__bf16*)(ws + 7 * MB + 192 * 1024);
  int*   kseg = (int*)   (ws + 7 * MB + 256 * 1024);

  const int nAct4 = NQc * DIMc / 4;   // 131072
  const int nW4   = DIMc * DIMc / 4;  // 4096
  cvt_kernel<<<(nAct4 + 255) / 256, 256, 0, stream>>>(query, Xq, nAct4);
  cvt_kernel<<<(nAct4 + 255) / 256, 256, 0, stream>>>(key,   Xk, nAct4);
  cvt_kernel<<<(nAct4 + 255) / 256, 256, 0, stream>>>(value, Xv, nAct4);
  cvt_kernel<<<(nW4 + 255) / 256, 256, 0, stream>>>(Wq, Wqb, nW4);
  cvt_kernel<<<(nW4 + 255) / 256, 256, 0, stream>>>(Wk, Wkb, nW4);
  cvt_kernel<<<(nW4 + 255) / 256, 256, 0, stream>>>(Wv, Wvb, nW4);
  cvt_kernel<<<(nW4 + 255) / 256, 256, 0, stream>>>(Wo, Wob, nW4);

  seg_kernel<<<(NKVc + 255) / 256, 256, 0, stream>>>(kvb, kseg, NKVc, NBc);

  gemm128_wmma<0><<<NQc / 256, 256, 0, stream>>>(Xq, Wqb, bq, qp, 0);
  gemm128_wmma<0><<<NKVc / 256, 256, 0, stream>>>(Xk, Wkb, bk, kp, 0);
  gemm128_wmma<1><<<NKVc / 256, 256, 0, stream>>>(Xv, Wvb, bv, vpT, NKVc);
  attn_wmma<<<NQc / 16, 128, 0, stream>>>(qp, kp, vpT, qb, kseg, kvb, atb);
  gemm128_wmma<2><<<NQc / 256, 256, 0, stream>>>(atb, Wob, bo, d_out, 0);
}
